// NMTDecoder_5566277616390
// MI455X (gfx1250) — matrix operand order repaired and checked by reading, not compiled, not run on hardware
//
#include <hip/hip_runtime.h>
#include <math.h>

// ---------------------------------------------------------------------------
// NMT GRU decoder for gfx1250 (MI455X), bf16 WMMA formulation.
//   V=32000 E=512 H=1024 B=32 T=48
// Phases:
//   1) convert W_ih/W_hh/W_cls f32 -> bf16 (into ws)
//   2) gather xs[t*B+b] = bf16(emb[tgt[b,t]])  (teacher forcing: all known)
//   3) Gi = xs @ W_ih^T  (1536x3072 bf16 WMMA GEMM)
//   4) 48 sequential gru_step kernels: gh = h @ W_hh^T via WMMA + fused gates
//   5) scores = Hs @ W_cls^T + b_cls as ONE GEMM (1536x32000) -> [B,T,V]
// GEMM staging uses CDNA5 async global->LDS DMA (ASYNCcnt) with LDS double
// buffering when the toolchain exposes the builtins; falls back to sync copy.
// ---------------------------------------------------------------------------

typedef __attribute__((ext_vector_type(16))) __bf16 v16bf;
typedef __attribute__((ext_vector_type(8)))  float  v8f;
typedef __attribute__((ext_vector_type(4)))  int    v4i;

typedef __attribute__((address_space(1))) v4i* gv4i_p;   // global int4*
typedef __attribute__((address_space(3))) v4i* lv4i_p;   // LDS int4*

#define LDSW   40            // padded LDS row stride (ushorts): 80B rows, conflict-free
#define TILE_E (128 * LDSW)  // elements per 128x32 staged tile buffer

#if defined(__has_builtin)
#  if __has_builtin(__builtin_amdgcn_global_load_async_to_lds_b128)
#    define HAVE_ASYNC_LDS 1
#  endif
#endif
#ifndef HAVE_ASYNC_LDS
#  define HAVE_ASYNC_LDS 0
#endif

__device__ __forceinline__ unsigned short f32_to_bf16_rne(float f) {
    union { float f; unsigned u; } v; v.f = f;
    unsigned u = v.u;
    unsigned r = u + 0x7FFFu + ((u >> 16) & 1u);
    return (unsigned short)(r >> 16);
}

__device__ __forceinline__ float sigmoidf_(float x) {
    return 1.0f / (1.0f + __expf(-x));
}

__device__ __forceinline__ void wait_async() {
#if HAVE_ASYNC_LDS
#  if __has_builtin(__builtin_amdgcn_s_wait_asynccnt)
    __builtin_amdgcn_s_wait_asynccnt(0);
#  else
    asm volatile("s_wait_asynccnt 0x0" ::: "memory");
#  endif
#endif
}

// ---- fragment loaders (contiguous k = 16*(lane/16) + s in A and B, wave32) -------------------------

// A fragment (16x32 bf16) from padded LDS tile As[rows][LDSW], rows m0..m0+15.
// lane<16: M=m0+lane, K=0..15; lane>=16: M=m0+lane-16, K=16..31
__device__ __forceinline__ v16bf frag_a_lds(const unsigned short* As, int lane, int m0) {
    int row = m0 + (lane & 15);
    int klo = (lane < 16) ? 0 : 16;
    union { uint4 q[2]; v16bf v; } f;
    f.q[0] = *(const uint4*)(As + row * LDSW + klo);
    f.q[1] = *(const uint4*)(As + row * LDSW + klo + 8);
    return f.v;
}

// B fragment (32x16 bf16) from padded LDS tile Bt[n][k] (K contiguous per n-row).
// lane<16: N=n0+lane, K=0..15; lane>=16: N=n0+lane-16, K=16..31
__device__ __forceinline__ v16bf frag_b_lds(const unsigned short* Bs, int lane, int n0) {
    int row = n0 + (lane & 15);
    int kk  = (lane < 16) ? 0 : 16;
    union { uint4 q[2]; v16bf v; } f;
    f.q[0] = *(const uint4*)(Bs + row * LDSW + kk);
    f.q[1] = *(const uint4*)(Bs + row * LDSW + kk + 8);
    return f.v;
}

// B fragment directly from a global bf16 weight matrix W[n][k], row stride ld.
__device__ __forceinline__ v16bf frag_b_glb(const unsigned short* W, long ld,
                                            int lane, long n0, int k0) {
    long row = n0 + (lane & 15);
    int  kk  = k0 + ((lane < 16) ? 0 : 16);
    union { uint4 q[2]; v16bf v; } f;
    const uint4* p = (const uint4*)(W + row * ld + kk);
    f.q[0] = p[0];
    f.q[1] = p[1];
    return f.v;
}

// A fragment from a global f32 matrix (h state, ld=1024), converted to bf16.
__device__ __forceinline__ v16bf frag_a_f32(const float* h, int lane, int m0, int k0) {
    int row = m0 + (lane & 15);
    int klo = k0 + ((lane < 16) ? 0 : 16);
    const float* p = h + row * 1024 + klo;
    union { unsigned short s[16]; v16bf v; } f;
#pragma unroll
    for (int e = 0; e < 8; ++e) f.s[e] = f32_to_bf16_rne(p[e]);
#pragma unroll
    for (int e = 0; e < 8; ++e) f.s[8 + e] = f32_to_bf16_rne(p[8 + e]);
    return f.v;
}

// stage one 128x32 bf16 tile: global (row stride ld) -> padded LDS, 256 threads.
// Async DMA to LDS on CDNA5 (2x b128 per thread), sync fallback otherwise.
__device__ __forceinline__ void stage_tile(unsigned short* dst, const unsigned short* src,
                                           long ld, long row0, int k0, int tid) {
    int r = tid >> 1, h = tid & 1;
    const unsigned short* s = src + (row0 + r) * ld + k0 + h * 16;
    unsigned short* d = dst + r * LDSW + h * 16;
#if HAVE_ASYNC_LDS
    __builtin_amdgcn_global_load_async_to_lds_b128(
        (gv4i_p)(void*)(unsigned short*)s, (lv4i_p)(void*)d, 0, 0);
    __builtin_amdgcn_global_load_async_to_lds_b128(
        (gv4i_p)(void*)(unsigned short*)(s + 8), (lv4i_p)(void*)(d + 8), 0, 0);
#else
    const uint4* sq = (const uint4*)s;
    uint4* dq = (uint4*)d;
    dq[0] = sq[0];
    dq[1] = sq[1];
#endif
}

// 128x128 block GEMM core, LDS double-buffered: 8 waves (2M x 4N),
// each wave 64x32 = 8 C tiles of 16x16. One barrier per K-tile.
__device__ __forceinline__ void gemm_tiles(const unsigned short* A, long lda, long rowA0,
                                           const unsigned short* B, long ldb, long rowB0,
                                           int ktiles, v8f acc[4][2],
                                           unsigned short* As, unsigned short* Bs) {
    int tid  = threadIdx.x;
    int lane = tid & 31;
    int wid  = tid >> 5;
    int wm   = wid >> 2;   // 0..1
    int wn   = wid & 3;    // 0..3

    stage_tile(As, A, lda, rowA0, 0, tid);
    stage_tile(Bs, B, ldb, rowB0, 0, tid);
    wait_async();
    __syncthreads();

    int p = 0;
    for (int kt = 0; kt < ktiles; ++kt) {
        int pn = p ^ 1;
        if (kt + 1 < ktiles) {   // prefetch next K-tile into the other buffer
            stage_tile(As + pn * TILE_E, A, lda, rowA0, (kt + 1) * 32, tid);
            stage_tile(Bs + pn * TILE_E, B, ldb, rowB0, (kt + 1) * 32, tid);
        }
        const unsigned short* Ac = As + p * TILE_E;
        const unsigned short* Bc = Bs + p * TILE_E;
        v16bf af[4];
#pragma unroll
        for (int mi = 0; mi < 4; ++mi) af[mi] = frag_a_lds(Ac, lane, wm * 64 + mi * 16);
#pragma unroll
        for (int nj = 0; nj < 2; ++nj) {
            v16bf bf = frag_b_lds(Bc, lane, wn * 32 + nj * 16);
#pragma unroll
            for (int mi = 0; mi < 4; ++mi) {
                acc[mi][nj] = __builtin_amdgcn_wmma_f32_16x16x32_bf16(
                    false, af[mi], false, bf, (short)0, acc[mi][nj], false, false);
            }
        }
        wait_async();
        __syncthreads();
        p = pn;
    }
}

// ---- phase kernels ---------------------------------------------------------

__global__ void cvt_f32_bf16(const float* __restrict__ src,
                             unsigned short* __restrict__ dst, int n) {
    int i = blockIdx.x * blockDim.x + threadIdx.x;
    int stride = gridDim.x * blockDim.x;
    for (; i < n; i += stride) dst[i] = f32_to_bf16_rne(src[i]);
}

__global__ void copy_h(const float* __restrict__ src, float* __restrict__ dst, int n) {
    int i = blockIdx.x * blockDim.x + threadIdx.x;
    if (i < n) dst[i] = src[i];
}

// xs[(t*32+b)][e] = bf16(emb[tgt[b][t]][e])
__global__ void gather_x(const float* __restrict__ emb, const int* __restrict__ tgt,
                         unsigned short* __restrict__ xs) {
    int m = blockIdx.x;        // 0..1535
    int t = m >> 5;            // m / 32
    int b = m & 31;
    long row = tgt[b * 48 + t];
    const float* s = emb + row * 512;
    unsigned short* d = xs + (long)m * 512;
    for (int e = threadIdx.x; e < 512; e += blockDim.x) d[e] = f32_to_bf16_rne(s[e]);
}

// Gi[1536 x 3072] = xs[1536 x 512] @ W_ih[3072 x 512]^T  (raw, biases in step)
__global__ void __launch_bounds__(256)
gemm_gi(const unsigned short* __restrict__ xs, const unsigned short* __restrict__ Wih,
        float* __restrict__ gi) {
    __shared__ unsigned short As[2 * TILE_E];
    __shared__ unsigned short Bs[2 * TILE_E];
    v8f acc[4][2] = {};
    long rowA0 = (long)blockIdx.y * 128;
    long rowB0 = (long)blockIdx.x * 128;
    gemm_tiles(xs, 512, rowA0, Wih, 512, rowB0, 16, acc, As, Bs);
    int lane = threadIdx.x & 31;
    int wid  = threadIdx.x >> 5;
    int wm = wid >> 2, wn = wid & 3;
#pragma unroll
    for (int nj = 0; nj < 2; ++nj) {
        int n = blockIdx.x * 128 + wn * 32 + nj * 16 + (lane & 15);
#pragma unroll
        for (int mi = 0; mi < 4; ++mi) {
            long mb = rowA0 + wm * 64 + mi * 16;
#pragma unroll
            for (int r = 0; r < 8; ++r) {
                long m = mb + ((lane < 16) ? r : r + 8);
                gi[m * 3072 + n] = acc[mi][nj][r];
            }
        }
    }
}

// one recurrence step: gh = h_old @ W_hh^T (WMMA), fused GRU gates
__global__ void __launch_bounds__(128)
gru_step(const float* __restrict__ h_old, float* __restrict__ h_new,
         const unsigned short* __restrict__ Whh, const float* __restrict__ gi,
         const float* __restrict__ b_ih, const float* __restrict__ b_hh,
         unsigned short* __restrict__ hs, int t) {
    int lane = threadIdx.x & 31;
    int wid  = threadIdx.x >> 5;              // 0..3 -> 16 columns each
    int j0   = blockIdx.x * 64 + wid * 16;    // hidden column base (0..1023)
    v8f ar[2] = {}, az[2] = {}, an[2] = {};
    for (int kt = 0; kt < 32; ++kt) {
        int k0 = kt * 32;
        v16bf a0 = frag_a_f32(h_old, lane, 0,  k0);
        v16bf a1 = frag_a_f32(h_old, lane, 16, k0);
        v16bf br = frag_b_glb(Whh, 1024, lane, 0 * 1024 + j0, k0);
        v16bf bz = frag_b_glb(Whh, 1024, lane, 1 * 1024 + j0, k0);
        v16bf bn = frag_b_glb(Whh, 1024, lane, 2 * 1024 + j0, k0);
        ar[0] = __builtin_amdgcn_wmma_f32_16x16x32_bf16(false, a0, false, br, (short)0, ar[0], false, false);
        ar[1] = __builtin_amdgcn_wmma_f32_16x16x32_bf16(false, a1, false, br, (short)0, ar[1], false, false);
        az[0] = __builtin_amdgcn_wmma_f32_16x16x32_bf16(false, a0, false, bz, (short)0, az[0], false, false);
        az[1] = __builtin_amdgcn_wmma_f32_16x16x32_bf16(false, a1, false, bz, (short)0, az[1], false, false);
        an[0] = __builtin_amdgcn_wmma_f32_16x16x32_bf16(false, a0, false, bn, (short)0, an[0], false, false);
        an[1] = __builtin_amdgcn_wmma_f32_16x16x32_bf16(false, a1, false, bn, (short)0, an[1], false, false);
    }
    int j = j0 + (lane & 15);
    float bihr = b_ih[j],        bhhr = b_hh[j];
    float bihz = b_ih[1024 + j], bhhz = b_hh[1024 + j];
    float bihn = b_ih[2048 + j], bhhn = b_hh[2048 + j];
#pragma unroll
    for (int ms = 0; ms < 2; ++ms) {
#pragma unroll
        for (int r = 0; r < 8; ++r) {
            int row = ms * 16 + ((lane < 16) ? r : r + 8);
            long gm = (long)(t * 32 + row) * 3072;
            float gir = gi[gm + j];
            float giz = gi[gm + 1024 + j];
            float gin = gi[gm + 2048 + j];
            float hv  = h_old[row * 1024 + j];
            float rg  = sigmoidf_(gir + bihr + ar[ms][r] + bhhr);
            float zg  = sigmoidf_(giz + bihz + az[ms][r] + bhhz);
            float ng  = tanhf(gin + bihn + rg * (an[ms][r] + bhhn));
            float hn  = (1.0f - zg) * ng + zg * hv;
            h_new[row * 1024 + j] = hn;
            hs[(long)(t * 32 + row) * 1024 + j] = f32_to_bf16_rne(hn);
        }
    }
}

// out[b][t][v] = Hs[t*32+b] @ W_cls^T + b_cls ; M=1536, N=32000, K=1024
__global__ void __launch_bounds__(256)
gemm_cls(const unsigned short* __restrict__ hs, const unsigned short* __restrict__ Wcls,
         const float* __restrict__ b_cls, float* __restrict__ out) {
    __shared__ unsigned short As[2 * TILE_E];
    __shared__ unsigned short Bs[2 * TILE_E];
    v8f acc[4][2] = {};
    long rowA0 = (long)blockIdx.y * 128;
    long rowB0 = (long)blockIdx.x * 128;
    gemm_tiles(hs, 1024, rowA0, Wcls, 1024, rowB0, 32, acc, As, Bs);
    int lane = threadIdx.x & 31;
    int wid  = threadIdx.x >> 5;
    int wm = wid >> 2, wn = wid & 3;
#pragma unroll
    for (int nj = 0; nj < 2; ++nj) {
        int n = blockIdx.x * 128 + wn * 32 + nj * 16 + (lane & 15);
        float bc = b_cls[n];
#pragma unroll
        for (int mi = 0; mi < 4; ++mi) {
            long mb = rowA0 + wm * 64 + mi * 16;
#pragma unroll
            for (int r = 0; r < 8; ++r) {
                long m  = mb + ((lane < 16) ? r : r + 8);
                long tt = m >> 5;   // timestep
                long bb = m & 31;   // batch
                out[(bb * 48 + tt) * 32000 + n] = acc[mi][nj][r] + bc;
            }
        }
    }
}

// ---------------------------------------------------------------------------

extern "C" void kernel_launch(void* const* d_in, const int* in_sizes, int n_in,
                              void* d_out, int out_size, void* d_ws, size_t ws_size,
                              hipStream_t stream) {
    const float* h0_in  = (const float*)d_in[0];   // [32,1024]
    const int*   tgt    = (const int*)d_in[1];     // [32,48]
    const float* emb    = (const float*)d_in[2];   // [32000,512]
    const float* Wih_f  = (const float*)d_in[3];   // [3072,512]
    const float* Whh_f  = (const float*)d_in[4];   // [3072,1024]
    const float* b_ih   = (const float*)d_in[5];   // [3072]
    const float* b_hh   = (const float*)d_in[6];   // [3072]
    const float* Wcls_f = (const float*)d_in[7];   // [32000,1024]
    const float* b_cls  = (const float*)d_in[8];   // [32000]
    float* out = (float*)d_out;                    // [32,48,32000]

    char* ws = (char*)d_ws;
    size_t off = 0;
    auto take = [&](size_t bytes) -> char* {
        char* p = ws + off;
        off = (off + bytes + 255) & ~(size_t)255;
        return p;
    };
    unsigned short* xs    = (unsigned short*)take(1536UL * 512 * 2);    // bf16 embeddings
    unsigned short* Wih   = (unsigned short*)take(3072UL * 512 * 2);
    unsigned short* Whh   = (unsigned short*)take(3072UL * 1024 * 2);
    unsigned short* Wcls  = (unsigned short*)take(32000UL * 1024 * 2);
    float*          giw   = (float*)take(1536UL * 3072 * 4);            // precomputed gi
    unsigned short* hsb   = (unsigned short*)take(1536UL * 1024 * 2);   // bf16 hidden states
    float*          hbuf0 = (float*)take(32UL * 1024 * 4);
    float*          hbuf1 = (float*)take(32UL * 1024 * 4);

    // 1) weight conversion f32 -> bf16
    cvt_f32_bf16<<<2048, 256, 0, stream>>>(Wih_f, Wih, 3072 * 512);
    cvt_f32_bf16<<<2048, 256, 0, stream>>>(Whh_f, Whh, 3072 * 1024);
    cvt_f32_bf16<<<4096, 256, 0, stream>>>(Wcls_f, Wcls, 32000 * 1024);

    // 2) gather all teacher-forced embeddings + init h
    gather_x<<<1536, 128, 0, stream>>>(emb, tgt, xs);
    copy_h<<<128, 256, 0, stream>>>(h0_in, hbuf0, 32 * 1024);

    // 3) Gi = xs @ W_ih^T for all timesteps at once
    gemm_gi<<<dim3(24, 12), 256, 0, stream>>>(xs, Wih, giw);

    // 4) sequential GRU recurrence (ping-pong h buffers)
    float* hA = hbuf0;
    float* hB = hbuf1;
    for (int t = 0; t < 48; ++t) {
        gru_step<<<16, 128, 0, stream>>>(hA, hB, Whh, giw, b_ih, b_hh, hsb, t);
        float* tmp = hA; hA = hB; hB = tmp;
    }

    // 5) classifier GEMM over all (t,b) rows, fused bias + [B,T,V] remap
    gemm_cls<<<dim3(250, 12), 256, 0, stream>>>(hsb, Wcls, b_cls, out);

    (void)in_sizes; (void)n_in; (void)out_size; (void)ws_size;
}
